// MultiHeadAttention_2594160247357
// MI455X (gfx1250) — compile-verified, matrix operand order checked by reading, not run on hardware
//
#include <hip/hip_runtime.h>
#include <math.h>

// Multi-head attention for MI455X (gfx1250), wave32 + WMMA bf16 + TDM.
// B=4, S=2048, D=1024, H=16, dh=64.

#define BB 4
#define SS 2048
#define DD 1024
#define HH 16
#define DH 64

typedef unsigned short u16;
typedef __attribute__((ext_vector_type(16))) __bf16 v16bf;
typedef __attribute__((ext_vector_type(8)))  float  v8f;
typedef __attribute__((ext_vector_type(4)))  unsigned int v4u;
typedef __attribute__((ext_vector_type(8)))  int    v8i;
typedef __attribute__((ext_vector_type(4)))  int    v4i;

union FragBF {
  v16bf v;
  u16   u[16];
  uint4 q[2];
};

// native f32 -> bf16 (v_cvt_bf16_f32 / packed), RNE
__device__ __forceinline__ u16 f2bf(float f) {
  union { __bf16 b; u16 u; } cv;
  cv.b = (__bf16)f;
  return cv.u;
}

#define WMMA_BF16(A, Bf, C) \
  __builtin_amdgcn_wmma_f32_16x16x32_bf16(false, (A), false, (Bf), (short)0, (C), false, false)

// 16-bit A-matrix fragment layout (ISA 7.12.2): lane = M (mod 16), half = lane/16.
// Elements 0..7  hold K = half*8 + 0..7       (contiguous -> one b128 load)
// Elements 8..15 hold K = 16 + half*8 + 0..7  (contiguous -> one b128 load)
__device__ __forceinline__ v16bf load_frag_bf16(const u16* row, int k0, int hx) {
  FragBF f;
  f.q[0] = *(const uint4*)(row + k0 + hx * 8);
  f.q[1] = *(const uint4*)(row + k0 + 16 + hx * 8);
  return f.v;
}

__device__ __forceinline__ v16bf load_frag_f32cvt(const float* row, int k0, int hx) {
  const float4* p0 = (const float4*)(row + k0 + hx * 8);
  const float4* p1 = (const float4*)(row + k0 + 16 + hx * 8);
  float4 a0 = p0[0], a1 = p0[1], b0 = p1[0], b1 = p1[1];
  v16bf f;
  f[0]  = (__bf16)a0.x; f[1]  = (__bf16)a0.y; f[2]  = (__bf16)a0.z; f[3]  = (__bf16)a0.w;
  f[4]  = (__bf16)a1.x; f[5]  = (__bf16)a1.y; f[6]  = (__bf16)a1.z; f[7]  = (__bf16)a1.w;
  f[8]  = (__bf16)b0.x; f[9]  = (__bf16)b0.y; f[10] = (__bf16)b0.z; f[11] = (__bf16)b0.w;
  f[12] = (__bf16)b1.x; f[13] = (__bf16)b1.y; f[14] = (__bf16)b1.z; f[15] = (__bf16)b1.w;
  return f;
}

// Row reductions across the 16 lanes holding one C-matrix row (xor 1/2/4/8 stays
// inside a 16-lane half, so the half->row mapping is preserved).
__device__ __forceinline__ float rmax16(float v) {
  v = fmaxf(v, __shfl_xor(v, 1, 32));
  v = fmaxf(v, __shfl_xor(v, 2, 32));
  v = fmaxf(v, __shfl_xor(v, 4, 32));
  v = fmaxf(v, __shfl_xor(v, 8, 32));
  return v;
}
__device__ __forceinline__ float rsum16(float v) {
  v += __shfl_xor(v, 1, 32);
  v += __shfl_xor(v, 2, 32);
  v += __shfl_xor(v, 4, 32);
  v += __shfl_xor(v, 8, 32);
  return v;
}

// Tensor Data Mover: DMA a 2-D bf16 tile (rows x DH, row stride DH elems) from
// global memory into LDS. D# fields per ISA cdna5 8.3/8.4.
__device__ __forceinline__ void tdm_load_tile(unsigned lds_off, const u16* gptr,
                                              unsigned rows) {
  unsigned long long ga = (unsigned long long)(const void*)gptr;
  v4u g0 = { 1u,                                   // count=1, no gather
             lds_off,                              // lds_addr (bytes)
             (unsigned)(ga & 0xffffffffull),       // global_addr[31:0]
             (unsigned)((ga >> 32) & 0x01ffffffu) | 0x80000000u }; // [56:32] | type=2
  v8i g1 = { (int)0x00010000u,                     // wg_mask=0, data_size=1 (2 bytes)
             (int)((unsigned)DH << 16),            // tensor_dim0[15:0] << 16
             (int)((unsigned)SS << 16),            // dim0_hi=0 | tensor_dim1[15:0]<<16
             (int)((unsigned)DH << 16),            // dim1_hi=0 | tile_dim0<<16
             (int)rows,                            // tile_dim1 | tile_dim2=0
             (int)DH,                              // tensor_dim0_stride[31:0]
             0, 0 };
  v4i gz = { 0, 0, 0, 0 };
#if __clang_major__ >= 23
  v8i gz8 = { 0, 0, 0, 0, 0, 0, 0, 0 };
  __builtin_amdgcn_tensor_load_to_lds(g0, g1, gz, gz, gz8, 0);
#else
  __builtin_amdgcn_tensor_load_to_lds(g0, g1, gz, gz, 0);
#endif
}

// ---------------------------------------------------------------- weights f32 -> bf16
__global__ void cvt_w(const float* __restrict__ wq, const float* __restrict__ wk,
                      const float* __restrict__ wv, const float* __restrict__ wo,
                      u16* __restrict__ dst) {
  int i = blockIdx.x * blockDim.x + threadIdx.x;     // 0 .. 4*D*D-1
  int m = i >> 20;                                    // D*D == 1<<20
  int j = i & ((1 << 20) - 1);
  const float* s = (m == 0) ? wq : (m == 1) ? wk : (m == 2) ? wv : wo;
  dst[i] = f2bf(s[j]);
}

// ---------------------------------------------------------------- Q/K/V projection
// Y = X @ W^T, written bf16 in head layout (B,H,S,dh).
// 4 waves/block; each wave computes a 16x64 strip (one head slice), so each
// converted A-fragment feeds 4 WMMAs.
__global__ void __launch_bounds__(128) proj_kernel(const float* __restrict__ X,
                                                   const u16* __restrict__ W,
                                                   u16* __restrict__ Oh) {
  const int lane = threadIdx.x & 31, hx = lane >> 4, l15 = lane & 15;
  const int w = threadIdx.x >> 5;
  const int s0 = blockIdx.x * 64 + w * 16;
  const int e0 = blockIdx.y * 64;                   // == head * DH
  const int b  = blockIdx.z;

  const float* arow = X + (size_t)(b * SS + s0 + l15) * DD;
  const u16* brow0 = W + (size_t)(e0 + l15) * DD;

  v8f c[4] = {};
  for (int k0 = 0; k0 < DD; k0 += 32) {
    __builtin_prefetch(arow + k0 + 128, 0, 0);
    v16bf a = load_frag_f32cvt(arow, k0, hx);
#pragma unroll
    for (int t = 0; t < 4; ++t) {
      v16bf wf = load_frag_bf16(brow0 + (size_t)t * 16 * DD, k0, hx);
      c[t] = WMMA_BF16(a, wf, c[t]);
    }
  }

  const int hh = blockIdx.y;
  u16* ob = Oh + ((size_t)(b * HH + hh) * SS + s0) * DH;
#pragma unroll
  for (int r = 0; r < 8; ++r) {
    const int row = r + 8 * hx;
#pragma unroll
    for (int t = 0; t < 4; ++t)
      ob[(size_t)row * DH + t * 16 + l15] = f2bf(c[t][r]);
  }
}

// ---------------------------------------------------------------- flash attention
// Block = 4 waves x 4 query tiles of one (b,h). The 32x64 V tile is DMA'd into
// LDS once per block by the Tensor Data Mover and shared by all 4 waves.
__global__ void __launch_bounds__(128) flash_kernel(const u16* __restrict__ Qh,
                                                    const u16* __restrict__ Kh,
                                                    const u16* __restrict__ Vh,
                                                    const int* __restrict__ valid_lens,
                                                    u16* __restrict__ Ctx) {
  __shared__ __align__(16) u16 vbuf[32 * DH];       // 4 KB shared V tile
  __shared__ __align__(16) u16 pbuf[4][16 * 32];    // per-wave P transpose buffers

  const int lane = threadIdx.x & 31, hx = lane >> 4, l15 = lane & 15;
  const int w  = threadIdx.x >> 5;
  const int q0 = blockIdx.x * 64 + w * 16;
  const int hh = blockIdx.y, b = blockIdx.z;
  const size_t bh = (size_t)(b * HH + hh) * SS;

  const u16* qrow = Qh + (bh + q0 + l15) * DH;
  v16bf aq0 = load_frag_bf16(qrow, 0, hx);
  v16bf aq1 = load_frag_bf16(qrow, 32, hx);

  v8f o[4] = {};
  float m[8], l[8];
#pragma unroll
  for (int r = 0; r < 8; ++r) { m[r] = -3.0e38f; l[r] = 0.0f; }

  const int      valid = valid_lens[b];
  const float    scale = 0.125f;                    // 1/sqrt(dh)
  const u16*     vbase = Vh + bh * DH;
  const unsigned vlds  = (unsigned)(unsigned long long)(const void*)&vbuf[0];
  u16* pmine = &pbuf[w][0];

  for (int kk = 0; kk < SS; kk += 32) {
    // ---- scores: two 16x16 tiles, K-dim = dh = 64 -> 2 WMMAs each
    const u16* krow0 = Kh + (bh + kk + l15) * DH;
    const u16* krow1 = krow0 + 16 * DH;
    v8f s0 = {}, s1 = {};
    {
      v16bf bk;
      bk = load_frag_bf16(krow0, 0,  hx); s0 = WMMA_BF16(aq0, bk, s0);
      bk = load_frag_bf16(krow0, 32, hx); s0 = WMMA_BF16(aq1, bk, s0);
      bk = load_frag_bf16(krow1, 0,  hx); s1 = WMMA_BF16(aq0, bk, s1);
      bk = load_frag_bf16(krow1, 32, hx); s1 = WMMA_BF16(aq1, bk, s1);
    }

    // ---- stage V tile via Tensor Data Mover (wave 0 issues for the block)
    __syncthreads();                                // prior vbuf reads complete
    if (w == 0) {
      asm volatile("s_wait_dscnt 0x0" ::: "memory");// WAR: LDS reads vs TDM write
      tdm_load_tile(vlds, vbase + (size_t)kk * DH, 32u);
      __builtin_amdgcn_s_wait_tensorcnt(0);
    }
    __syncthreads();                                // vbuf ready for all waves

    // ---- online softmax update (per C-layout row), P -> per-wave LDS
    const int c0 = kk + l15, c1 = c0 + 16;
#pragma unroll
    for (int r = 0; r < 8; ++r) {
      float v0 = s0[r] * scale;
      float v1 = s1[r] * scale;
      if (c0 >= valid) v0 = -1.0e6f;                // sequence_mask semantics
      if (c1 >= valid) v1 = -1.0e6f;
      float mn    = rmax16(fmaxf(v0, v1));
      float mi    = fmaxf(m[r], mn);
      float alpha = __expf(m[r] - mi);
      float p0    = __expf(v0 - mi);
      float p1    = __expf(v1 - mi);
      l[r] = l[r] * alpha + rsum16(p0 + p1);
      m[r] = mi;
      o[0][r] = o[0][r] * alpha;
      o[1][r] = o[1][r] * alpha;
      o[2][r] = o[2][r] * alpha;
      o[3][r] = o[3][r] * alpha;
      const int row = r + 8 * hx;                   // C-layout row for this lane/VGPR
      pmine[row * 32 + l15]      = f2bf(p0);
      pmine[row * 32 + 16 + l15] = f2bf(p1);
    }

    // P: C-layout -> A-layout (same-wave DS ops are in-order)
    v16bf pa = load_frag_bf16(pmine + l15 * 32, 0, hx);

    // ---- O += P (16x32) * V (32 x dh); V B-fragments gathered from LDS tile
#pragma unroll
    for (int t = 0; t < 4; ++t) {
      FragBF bv;
      const int dcol = t * 16 + l15;
#pragma unroll
      for (int j = 0; j < 8; ++j) {
        bv.u[j]     = vbuf[(hx * 8 + j) * DH + dcol];
        bv.u[8 + j] = vbuf[(16 + hx * 8 + j) * DH + dcol];
      }
      o[t] = WMMA_BF16(pa, bv.v, o[t]);
    }
  }

  // normalize and write context (B,S,D) bf16
  u16* cb = Ctx + ((size_t)(b * SS) + q0) * DD + hh * DH;
#pragma unroll
  for (int r = 0; r < 8; ++r) {
    const float inv = 1.0f / l[r];
    const int   row = r + 8 * hx;
#pragma unroll
    for (int t = 0; t < 4; ++t)
      cb[(size_t)row * DD + t * 16 + l15] = f2bf(o[t][r] * inv);
  }
}

// ---------------------------------------------------------------- output projection
// Y = Ctx(bf16) @ Wo^T -> f32. 4 waves/block, each wave a 16x64 strip.
__global__ void __launch_bounds__(128) oproj_kernel(const u16* __restrict__ Ctx,
                                                    const u16* __restrict__ W,
                                                    float* __restrict__ Y) {
  const int lane = threadIdx.x & 31, hx = lane >> 4, l15 = lane & 15;
  const int w = threadIdx.x >> 5;
  const int s0 = blockIdx.x * 64 + w * 16;
  const int e0 = blockIdx.y * 64;
  const int b  = blockIdx.z;

  const u16* arow  = Ctx + (size_t)(b * SS + s0 + l15) * DD;
  const u16* brow0 = W + (size_t)(e0 + l15) * DD;

  v8f c[4] = {};
  for (int k0 = 0; k0 < DD; k0 += 32) {
    __builtin_prefetch(arow + k0 + 128, 0, 0);
    v16bf a = load_frag_bf16(arow, k0, hx);
#pragma unroll
    for (int t = 0; t < 4; ++t) {
      v16bf wf = load_frag_bf16(brow0 + (size_t)t * 16 * DD, k0, hx);
      c[t] = WMMA_BF16(a, wf, c[t]);
    }
  }

  float* yb = Y + ((size_t)(b * SS) + s0) * DD + e0;
#pragma unroll
  for (int r = 0; r < 8; ++r) {
    const int row = r + 8 * hx;
#pragma unroll
    for (int t = 0; t < 4; ++t)
      yb[(size_t)row * DD + t * 16 + l15] = c[t][r];
  }
}

// ---------------------------------------------------------------- launch
extern "C" void kernel_launch(void* const* d_in, const int* in_sizes, int n_in,
                              void* d_out, int out_size, void* d_ws, size_t ws_size,
                              hipStream_t stream) {
  const float* q  = (const float*)d_in[0];
  const float* k  = (const float*)d_in[1];
  const float* v  = (const float*)d_in[2];
  const int*   vl = (const int*)d_in[3];
  const float* wq = (const float*)d_in[4];
  const float* wk = (const float*)d_in[5];
  const float* wv = (const float*)d_in[6];
  const float* wo = (const float*)d_in[7];

  // workspace layout (bf16): Wq|Wk|Wv|Wo (4*1M) + Qh|Kh|Vh|Ctx (4*8.39M) = 72 MB
  u16* ws  = (u16*)d_ws;
  u16* Wqb = ws;
  u16* Wkb = Wqb + (size_t)DD * DD;
  u16* Wvb = Wkb + (size_t)DD * DD;
  u16* Wob = Wvb + (size_t)DD * DD;
  const size_t HE = (size_t)BB * HH * SS * DH;
  u16* Qh  = Wob + (size_t)DD * DD;
  u16* Kh  = Qh + HE;
  u16* Vh  = Kh + HE;
  u16* Ctx = Vh + HE;

  cvt_w<<<(4 * DD * DD) / 256, 256, 0, stream>>>(wq, wk, wv, wo, ws);

  dim3 gp(SS / 64, DD / 64, BB);
  proj_kernel<<<gp, 128, 0, stream>>>(q, Wqb, Qh);
  proj_kernel<<<gp, 128, 0, stream>>>(k, Wkb, Kh);
  proj_kernel<<<gp, 128, 0, stream>>>(v, Wvb, Vh);

  flash_kernel<<<dim3(SS / 64, HH, BB), 128, 0, stream>>>(Qh, Kh, Vh, vl, Ctx);

  oproj_kernel<<<gp, 128, 0, stream>>>(Ctx, Wob, (float*)d_out);
}
